// SPECrossAtt_8787503088256
// MI455X (gfx1250) — matrix operand order repaired and checked by reading, not compiled, not run on hardware
//
#include <hip/hip_runtime.h>
#include <hip/hip_bf16.h>
#include <cstdint>

// ---------------------------------------------------------------------------
// Problem constants: B=16, C=256, H=W=64, N=4096
// ---------------------------------------------------------------------------
#define BATCH 16
#define CDIM  256
#define NDIM  4096

typedef __bf16 v16bf __attribute__((ext_vector_type(16)));
typedef __bf16 v2bf  __attribute__((ext_vector_type(2)));
typedef float  v8f   __attribute__((ext_vector_type(8)));
typedef float  v4f   __attribute__((ext_vector_type(4)));
typedef unsigned int u32x4 __attribute__((ext_vector_type(4)));

static constexpr int BKc = 32;   // WMMA K per instruction (bf16 16x16x32)
static constexpr int BKs = 64;   // staged K per LDS tile (= 2 WMMA steps/barrier)

// Hardware bf16 conversion (backend should select v_cvt_pk_bf16_f32).
__device__ inline unsigned int pack2(float lo, float hi) {
    v2bf v;
    v[0] = (__bf16)lo;
    v[1] = (__bf16)hi;
    return __builtin_bit_cast(unsigned int, v);
}

// A fragment (16x32 bf16): lane l -> row (l&15); K = two 8-wide runs at
// (l>>4)*8 and (l>>4)*8+16  (CDNA5 ISA 7.12.2, 16-bit A layout)
__device__ inline v16bf load_frag_a(const unsigned short* lds, int row, int kk, int lane) {
    const int base = row * BKs + kk * BKc + (lane >> 4) * 8;
    union { v16bf v; u32x4 q[2]; } u;
    u.q[0] = *(const u32x4*)&lds[base];
    u.q[1] = *(const u32x4*)&lds[base + 16];
    return u.v;
}

// B fragment (32x16 bf16): lane l -> col (l&15); K = two 8-wide runs at
// (l>>4)*8 and (l>>4)*8+16 (matches A)
__device__ inline v16bf load_frag_b(const unsigned short* lds, int row, int kk, int lane) {
    const int base = row * BKs + kk * BKc + (lane >> 4) * 8;
    union { v16bf v; u32x4 q[2]; } u;
    u.q[0] = *(const u32x4*)&lds[base];
    u.q[1] = *(const u32x4*)&lds[base + 16];
    return u.v;
}

// Stage a [ROWS x BKs] f32 tile (row-major, contiguous in K) into LDS as bf16.
template <int ROWS>
__device__ inline void stage_rowmajor(const float* __restrict__ src, int ld, int k0,
                                      unsigned short* lds, int tid) {
    constexpr int CH = ROWS * BKs / 256;         // floats per thread (contig in K)
    const int start = tid * CH;
    const int r = start / BKs;
    const int koff = start % BKs;
    const float* s = src + (long long)r * ld + k0 + koff;
    unsigned int* d = (unsigned int*)(lds + r * BKs + koff);
    const v4f* sv = (const v4f*)s;
#pragma unroll
    for (int i = 0; i < CH / 4; ++i) {
        v4f v = sv[i];
        d[2 * i]     = pack2(v[0], v[1]);
        d[2 * i + 1] = pack2(v[2], v[3]);
    }
    __builtin_prefetch(s + 2 * BKs, 0, 1);       // global_prefetch, 2 tiles ahead
}

// Stage a [BKs x BN] f32 tile (K-major in memory) into LDS laid out [n][k].
// Each thread handles a k-pair so LDS stores are packed u32 (k, k+1).
template <int BN>
__device__ inline void stage_kmajor(const float* __restrict__ src, int ld,
                                    unsigned short* lds, int tid) {
    constexpr int PAIRS = BKs / 2;               // 32 k-pairs
    constexpr int TPN = 256 / PAIRS;             // threads along n per k-pair
    constexpr int CH = BN / TPN;                 // n-values per thread
    const int k2 = tid / TPN;                    // k-pair index: k = 2*k2, 2*k2+1
    const int nq = (tid % TPN) * CH;
    const float* s0 = src + (long long)(2 * k2) * ld + nq;
    const float* s1 = s0 + ld;
    const v4f* sv0 = (const v4f*)s0;
    const v4f* sv1 = (const v4f*)s1;
#pragma unroll
    for (int q = 0; q < CH / 4; ++q) {
        v4f a = sv0[q];
        v4f b = sv1[q];
#pragma unroll
        for (int i = 0; i < 4; ++i)
            *(unsigned int*)(lds + (nq + 4 * q + i) * BKs + 2 * k2) = pack2(a[i], b[i]);
    }
    __builtin_prefetch(s0 + (long long)2 * BKs * ld, 0, 1);
}

// ---------------------------------------------------------------------------
// Generic bf16 WMMA GEMM:  C[b] = A[b] (MxK) * B[b] (KxN), LDS double-buffered.
//   BT=true : B supplied as Bt (N x K, row-major)  -> A*Bt^T   (NT)
//   BT=false: B supplied as  B (K x N, row-major)               (NN)
//   EPI=0: C = acc
//   EPI=2: C = beta*(acc + u[m]) + resid[m,n]  (residual, non-temporal stores)
//   LDC  : compile-time output/resid leading dimension (immediate store offsets)
// 8 wave32s arranged 2 (M) x 4 (N); each wave owns (BM/2) x (BN/4).
// ---------------------------------------------------------------------------
template <int BM, int BN, bool BT, int EPI, int LDC>
__global__ __launch_bounds__(256) void gemm_bf16_kernel(
    const float* __restrict__ A, int lda, long long sA,
    const float* __restrict__ B, int ldb, long long sB,
    float* __restrict__ C, long long sC,
    int K,
    const float* __restrict__ resid, long long sR,
    const float* __restrict__ uvec, long long sU,
    const float* __restrict__ beta_ptr) {
    constexpr int MI = BM / 32;   // M sub-tiles per wave
    constexpr int NJ = BN / 64;   // N sub-tiles per wave

    __shared__ unsigned short Alds[2][BM * BKs];
    __shared__ unsigned short Blds[2][BN * BKs];

    const int batch = blockIdx.z;
    const float* Ab = A + (long long)batch * sA + (long long)(blockIdx.y * BM) * lda;
    const float* Bb = B + (long long)batch * sB;
    float* Cb = C + (long long)batch * sC;
    const int m0 = blockIdx.y * BM;
    const int n0 = blockIdx.x * BN;

    const int tid = threadIdx.x;
    const int lane = tid & 31;
    const int wid = tid >> 5;
    const int wm = (wid >> 2) * (BM / 2);
    const int wn = (wid & 3) * (BN / 4);

    v8f acc[MI][NJ];
#pragma unroll
    for (int i = 0; i < MI; ++i)
#pragma unroll
        for (int j = 0; j < NJ; ++j) {
            v8f z = {0.f, 0.f, 0.f, 0.f, 0.f, 0.f, 0.f, 0.f};
            acc[i][j] = z;
        }

    auto stageAB = [&](int buf, int k0) {
        stage_rowmajor<BM>(Ab, lda, k0, Alds[buf], tid);
        if constexpr (BT) {
            stage_rowmajor<BN>(Bb + (long long)n0 * ldb, ldb, k0, Blds[buf], tid);
        } else {
            stage_kmajor<BN>(Bb + (long long)k0 * ldb + n0, ldb, Blds[buf], tid);
        }
    };

    stageAB(0, 0);
    int buf = 0;
    for (int k0 = 0; k0 < K; k0 += BKs) {
        __syncthreads();                       // staged tile 'buf' now visible
        if (k0 + BKs < K) stageAB(buf ^ 1, k0 + BKs);   // overlap next stage

#pragma unroll
        for (int kk = 0; kk < BKs / BKc; ++kk) {
            v16bf af[MI], bfb[NJ];
#pragma unroll
            for (int i = 0; i < MI; ++i)
                af[i] = load_frag_a(Alds[buf], wm + i * 16 + (lane & 15), kk, lane);
#pragma unroll
            for (int j = 0; j < NJ; ++j)
                bfb[j] = load_frag_b(Blds[buf], wn + j * 16 + (lane & 15), kk, lane);

#pragma unroll
            for (int i = 0; i < MI; ++i)
#pragma unroll
                for (int j = 0; j < NJ; ++j)
                    acc[i][j] = __builtin_amdgcn_wmma_f32_16x16x32_bf16(
                        false, af[i], false, bfb[j], (short)0, acc[i][j], false, false);
        }
        buf ^= 1;
    }

    // Epilogue. C/D layout: elem e, lane l -> m = e + 8*(l>>4), n = (l&15).
    // LDC is a compile-time constant -> e*LDC folds into store immediates.
#pragma unroll
    for (int i = 0; i < MI; ++i)
#pragma unroll
        for (int j = 0; j < NJ; ++j) {
            const int tm = m0 + wm + i * 16 + (lane >> 4) * 8;
            const int tn = n0 + wn + j * 16 + (lane & 15);
            float* cbase = Cb + (long long)tm * LDC + tn;
            if constexpr (EPI == 2) {
                const float beta = beta_ptr[0];
                const float* rbase = resid + (long long)batch * sR + (long long)tm * LDC + tn;
                const float* ubase = uvec + (long long)batch * sU + tm;
#pragma unroll
                for (int e = 0; e < 8; ++e) {
                    const float v = beta * (acc[i][j][e] + ubase[e]) + rbase[e * LDC];
                    __builtin_nontemporal_store(v, &cbase[e * LDC]);   // write-once output
                }
            } else {
#pragma unroll
                for (int e = 0; e < 8; ++e)
                    cbase[e * LDC] = acc[i][j][e];
            }
        }
}

// ---------------------------------------------------------------------------
// rowsum: r[b][c] = sum_n X[b,c,n]   (one block per row; grid.x = 2*B*C)
// ---------------------------------------------------------------------------
__global__ __launch_bounds__(256) void rowsum_kernel(const float* __restrict__ x1,
                                                     const float* __restrict__ x2,
                                                     float* __restrict__ r1,
                                                     float* __restrict__ r2) {
    const int idx = blockIdx.x;
    const int which = idx >> 12;           // B*C = 4096
    const int bc = idx & 4095;
    const float* src = (which ? x2 : x1) + (long long)bc * NDIM;
    float* dst = which ? r2 : r1;
    float s = 0.f;
    for (int i = threadIdx.x; i < NDIM; i += 256) s += src[i];
    __shared__ float red[256];
    red[threadIdx.x] = s;
    __syncthreads();
    for (int off = 128; off > 0; off >>= 1) {
        if (threadIdx.x < off) red[threadIdx.x] += red[threadIdx.x + off];
        __syncthreads();
    }
    if (threadIdx.x == 0) dst[bc] = red[0];
}

// ---------------------------------------------------------------------------
// matvec: per batch b:
//   t=0: p1 = Wq2*r2 + N*bq2    t=1: g1 = Wk1*r1
//   t=2: p2 = Wq1*r1 + N*bq1    t=3: g2 = Wk2*r2
// grid (4, B), block 256 (thread = output row)
// ---------------------------------------------------------------------------
__global__ __launch_bounds__(256) void matvec_kernel(
    const float* __restrict__ wq1, const float* __restrict__ bq1,
    const float* __restrict__ wk1, const float* __restrict__ wq2,
    const float* __restrict__ bq2, const float* __restrict__ wk2,
    const float* __restrict__ r1, const float* __restrict__ r2,
    float* __restrict__ p1, float* __restrict__ g1,
    float* __restrict__ p2, float* __restrict__ g2) {
    const int b = blockIdx.y;
    const int t = blockIdx.x;
    const int c = threadIdx.x;
    const float* Wm;
    const float* v;
    const float* bias = nullptr;
    float* out;
    switch (t) {
        case 0: Wm = wq2; v = r2 + b * CDIM; bias = bq2; out = p1 + b * CDIM; break;
        case 1: Wm = wk1; v = r1 + b * CDIM;             out = g1 + b * CDIM; break;
        case 2: Wm = wq1; v = r1 + b * CDIM; bias = bq1; out = p2 + b * CDIM; break;
        default: Wm = wk2; v = r2 + b * CDIM;            out = g2 + b * CDIM; break;
    }
    float s = bias ? (float)NDIM * bias[c] : 0.f;
    for (int k = 0; k < CDIM; ++k) s += Wm[c * CDIM + k] * v[k];
    out[c] = s;
}

// ---------------------------------------------------------------------------
// softmax over rows of energy = E + p[c]*bk[d] + bq[c]*g[d]
//   attn[c,d] = softmax_d(rowmax - energy) = exp(rowmin - energy)/sum
// also u[c] = sum_d attn[c,d]*bv[d].  In-place on E.
// grid (C/8, B, 2); 8 wave32 per block, one row per wave (8 elems/lane).
// ---------------------------------------------------------------------------
__global__ __launch_bounds__(256) void softmax_kernel(
    float* __restrict__ E1, float* __restrict__ E2,
    const float* __restrict__ p1, const float* __restrict__ g1,
    const float* __restrict__ bk1, const float* __restrict__ bq2,
    const float* __restrict__ p2, const float* __restrict__ g2,
    const float* __restrict__ bk2, const float* __restrict__ bq1,
    const float* __restrict__ bv1, const float* __restrict__ bv2,
    float* __restrict__ u1, float* __restrict__ u2) {
    const int b = blockIdx.y;
    const int which = blockIdx.z;
    const int wid = threadIdx.x >> 5;
    const int lane = threadIdx.x & 31;
    const int c = blockIdx.x * 8 + wid;

    float* E = (which ? E2 : E1) + ((long long)b << 16) + (long long)c * CDIM;
    const float* p = (which ? p2 : p1) + b * CDIM;
    const float* g = (which ? g2 : g1) + b * CDIM;
    const float* bk = which ? bk2 : bk1;
    const float* bq = which ? bq1 : bq2;
    const float* bv = which ? bv2 : bv1;
    float* u = (which ? u2 : u1) + b * CDIM;

    const float pc = p[c];
    const float bqc = bq[c];
    float vals[8];
    float rmin = 3.0e38f;
#pragma unroll
    for (int j = 0; j < 8; ++j) {
        const int d = lane + 32 * j;
        const float v = E[d] + pc * bk[d] + bqc * g[d];
        vals[j] = v;
        rmin = fminf(rmin, v);
    }
    for (int off = 16; off > 0; off >>= 1)
        rmin = fminf(rmin, __shfl_xor(rmin, off, 32));

    float ex[8];
    float ssum = 0.f;
#pragma unroll
    for (int j = 0; j < 8; ++j) {
        ex[j] = expf(rmin - vals[j]);   // == exp(inv - max(inv))
        ssum += ex[j];
    }
    for (int off = 16; off > 0; off >>= 1)
        ssum += __shfl_xor(ssum, off, 32);
    const float inv = 1.0f / ssum;

    float su = 0.f;
#pragma unroll
    for (int j = 0; j < 8; ++j) {
        const int d = lane + 32 * j;
        const float a = ex[j] * inv;
        E[d] = a;
        su += a * bv[d];
    }
    for (int off = 16; off > 0; off >>= 1)
        su += __shfl_xor(su, off, 32);
    if (lane == 0) u[c] = su;
}

// ---------------------------------------------------------------------------
// Host-side orchestration
// ---------------------------------------------------------------------------
extern "C" void kernel_launch(void* const* d_in, const int* in_sizes, int n_in,
                              void* d_out, int out_size, void* d_ws, size_t ws_size,
                              hipStream_t stream) {
    (void)in_sizes; (void)n_in; (void)out_size; (void)ws_size;
    const float* input1 = (const float*)d_in[0];
    const float* input2 = (const float*)d_in[1];
    const float* wq1 = (const float*)d_in[2];
    const float* bq1 = (const float*)d_in[3];
    const float* wk1 = (const float*)d_in[4];
    const float* bk1 = (const float*)d_in[5];
    const float* wv1 = (const float*)d_in[6];
    const float* bv1 = (const float*)d_in[7];
    const float* wq2 = (const float*)d_in[8];
    const float* bq2 = (const float*)d_in[9];
    const float* wk2 = (const float*)d_in[10];
    const float* bk2 = (const float*)d_in[11];
    const float* wv2 = (const float*)d_in[12];
    const float* bv2 = (const float*)d_in[13];
    const float* beta = (const float*)d_in[14];
    float* out = (float*)d_out;

    // workspace layout (floats). Per-matrix block = B*C*C = 1,048,576 floats.
    const long long MAT = (long long)BATCH * CDIM * CDIM;
    float* ws = (float*)d_ws;
    float* S  = ws;                 // [B,C,C]  X2*X1^T
    float* T1 = ws + 1 * MAT;       // Wq2*S        -> later reused for M1
    float* T2 = ws + 2 * MAT;       // Wq1*S^T      -> later reused for M2
    float* E1 = ws + 3 * MAT;       // energy1 -> attn1 (in place)
    float* E2 = ws + 4 * MAT;       // energy2 -> attn2
    float* vecs = ws + 5 * MAT;     // 8 x [B,C]
    float* r1 = vecs + 0 * BATCH * CDIM;
    float* r2 = vecs + 1 * BATCH * CDIM;
    float* p1 = vecs + 2 * BATCH * CDIM;
    float* g1 = vecs + 3 * BATCH * CDIM;
    float* p2 = vecs + 4 * BATCH * CDIM;
    float* g2 = vecs + 5 * BATCH * CDIM;
    float* u1 = vecs + 6 * BATCH * CDIM;
    float* u2 = vecs + 7 * BATCH * CDIM;

    const long long sX = (long long)CDIM * NDIM;   // batch stride of inputs/outputs
    const long long sM = (long long)CDIM * CDIM;   // batch stride of CxC mats
    const long long sV = CDIM;                     // batch stride of vectors

    // 1) row sums of X1, X2
    rowsum_kernel<<<2 * BATCH * CDIM, 256, 0, stream>>>(input1, input2, r1, r2);

    // 2) bias-correction vectors
    matvec_kernel<<<dim3(4, BATCH), 256, 0, stream>>>(
        wq1, bq1, wk1, wq2, bq2, wk2, r1, r2, p1, g1, p2, g2);

    // 3) Gram: S = X2 * X1^T  (NT, K=4096), 64x64 tiles -> 256 blocks
    gemm_bf16_kernel<64, 64, true, 0, CDIM><<<dim3(4, 4, BATCH), 256, 0, stream>>>(
        input2, NDIM, sX, input1, NDIM, sX, S, sM, NDIM,
        nullptr, 0, nullptr, 0, nullptr);

    // 4) T1 = Wq2 * S (NN), T2 = Wq1 * S^T (NT); K=256
    gemm_bf16_kernel<128, 128, false, 0, CDIM><<<dim3(2, 2, BATCH), 256, 0, stream>>>(
        wq2, CDIM, 0, S, CDIM, sM, T1, sM, CDIM,
        nullptr, 0, nullptr, 0, nullptr);
    gemm_bf16_kernel<128, 128, true, 0, CDIM><<<dim3(2, 2, BATCH), 256, 0, stream>>>(
        wq1, CDIM, 0, S, CDIM, sM, T2, sM, CDIM,
        nullptr, 0, nullptr, 0, nullptr);

    // 5) E1 = T1 * Wk1^T, E2 = T2 * Wk2^T  (NT)
    gemm_bf16_kernel<128, 128, true, 0, CDIM><<<dim3(2, 2, BATCH), 256, 0, stream>>>(
        T1, CDIM, sM, wk1, CDIM, 0, E1, sM, CDIM,
        nullptr, 0, nullptr, 0, nullptr);
    gemm_bf16_kernel<128, 128, true, 0, CDIM><<<dim3(2, 2, BATCH), 256, 0, stream>>>(
        T2, CDIM, sM, wk2, CDIM, 0, E2, sM, CDIM,
        nullptr, 0, nullptr, 0, nullptr);

    // 6) softmax(rowmax - energy) with rank-1 corrections; u = attn*bv
    softmax_kernel<<<dim3(CDIM / 8, BATCH, 2), 256, 0, stream>>>(
        E1, E2, p1, g1, bk1, bq2, p2, g2, bk2, bq1, bv1, bv2, u1, u2);

    // 7) M1 = attn1 * Wv1 (NN, into T1), M2 = attn2 * Wv2 (into T2)
    gemm_bf16_kernel<128, 128, false, 0, CDIM><<<dim3(2, 2, BATCH), 256, 0, stream>>>(
        E1, CDIM, sM, wv1, CDIM, 0, T1, sM, CDIM,
        nullptr, 0, nullptr, 0, nullptr);
    gemm_bf16_kernel<128, 128, false, 0, CDIM><<<dim3(2, 2, BATCH), 256, 0, stream>>>(
        E2, CDIM, sM, wv2, CDIM, 0, T2, sM, CDIM,
        nullptr, 0, nullptr, 0, nullptr);

    // 8) outspe1 = beta*(M1*X1 + u1) + input1 ; outspe2 = beta*(M2*X2 + u2) + input2
    gemm_bf16_kernel<128, 128, false, 2, NDIM><<<dim3(NDIM / 128, 2, BATCH), 256, 0, stream>>>(
        T1, CDIM, sM, input1, NDIM, sX, out, sX, CDIM,
        input1, sX, u1, sV, beta);
    gemm_bf16_kernel<128, 128, false, 2, NDIM><<<dim3(NDIM / 128, 2, BATCH), 256, 0, stream>>>(
        T2, CDIM, sM, input2, NDIM, sX, out + (long long)BATCH * sX, sX, CDIM,
        input2, sX, u2, sV, beta);
}
